// GCNBiLSTM_5403068858446
// MI455X (gfx1250) — matrix operand order repaired and checked by reading, not compiled, not run on hardware
//
#include <hip/hip_runtime.h>
#include <hip/hip_bf16.h>
#include <stdint.h>
#include <stddef.h>

// ---------------- problem constants ----------------
#define B_      128
#define T_      512
#define H_      128
#define GCNH    32
#define NCLS    10
#define FIN     64
#define LSTM_IN 160
#define N_NODES (B_ * 5 * T_)      // 327680
#define E_EDGES 2097152

typedef __attribute__((ext_vector_type(16))) _Float16 v16h;
typedef __attribute__((ext_vector_type(8)))  float    v8f;
typedef __attribute__((ext_vector_type(4)))  unsigned int u32x4;
typedef __attribute__((ext_vector_type(8)))  int       i32x8;
typedef __attribute__((ext_vector_type(4)))  int       i32x4;

// ---------------- WMMA helpers (CDNA5 16x16x32 f16 -> f32) ----------------
__device__ __forceinline__ v8f wmma16(v16h a, v16h b, v8f c) {
  // (neg_a, A, neg_b, B, c_mod, C, reuse_a, reuse_b)
  return __builtin_amdgcn_wmma_f32_16x16x32_f16(false, a, false, b, (short)0, c,
                                                false, false);
}

// A fragment: 16x32 tile from f32 row-major matrix.
// lane L: row = L%16 ; half i -> K = k0 + (i/8)*16 + (L/16)*8 + (i%8)
template <bool RELU>
__device__ __forceinline__ v16h load_a_frag(const float* __restrict__ base,
                                            int row0, int lda, int k0) {
  int lane = threadIdx.x & 31;
  int r = lane & 15, hi = lane >> 4;
  const float* p = base + (size_t)(row0 + r) * lda + k0 + hi * 8;
  v16h a;
#pragma unroll
  for (int g = 0; g < 2; ++g) {
#pragma unroll
    for (int i = 0; i < 8; ++i) {
      float v = p[g * 16 + i];
      if (RELU) v = fmaxf(v, 0.0f);
      a[g * 8 + i] = (_Float16)v;
    }
  }
  return a;
}

// D fragment store: lane L -> col c = L%16, vgpr j -> row j + 8*(L/16)
__device__ __forceinline__ void store_d(float* __restrict__ base, int row0,
                                        int ldc, int col0, v8f acc) {
  int lane = threadIdx.x & 31;
  int c = lane & 15, hi = lane >> 4;
#pragma unroll
  for (int j = 0; j < 8; ++j)
    base[(size_t)(row0 + j + 8 * hi) * ldc + col0 + c] = acc[j];
}

__device__ __forceinline__ float sigmf(float x) {
  x = fminf(fmaxf(x, -30.f), 30.f);
  return 1.0f / (1.0f + __expf(-x));
}
__device__ __forceinline__ float tanh_f(float x) {
  x = fminf(fmaxf(x, -15.f), 15.f);
  float e = __expf(2.0f * x);
  return (e - 1.0f) / (e + 1.0f);
}

// -------- Tensor Data Mover: 2-D tile (16 rows x 512 f32) global -> LDS ----
// D# layout per CDNA5 ISA.  Rows strided by T*512 elems.
__device__ __forceinline__ void tdm_load_gate_tile(unsigned lds_byte_off,
                                                   const float* gptr) {
  unsigned long long ga = (unsigned long long)(uintptr_t)gptr;
  u32x4 g0;
  g0[0] = 1u;                                   // count=1 (valid), user mode
  g0[1] = lds_byte_off;                         // lds_addr (bytes)
  g0[2] = (unsigned)(ga & 0xffffffffull);       // global_addr[31:0]
  g0[3] = (unsigned)((ga >> 32) & 0x1ffffffull) // global_addr[56:32]
          | (2u << 30);                         // type=2 ("image")
  i32x8 g1;
  const unsigned td0 = 512u, td1 = 16u, tile0 = 512u, tile1 = 16u;
  const unsigned stride0 = (unsigned)(T_ * 512); // elements between rows
  g1[0] = (int)(2u << 16);                      // wg_mask=0, data_size=2 (4B)
  g1[1] = (int)((td0 & 0xffffu) << 16);         // tensor_dim0[15:0] @bit48
  g1[2] = (int)((td0 >> 16) & 0xffffu) | (int)((td1 & 0xffffu) << 16);
  g1[3] = (int)((td1 >> 16) & 0xffffu) | (int)(tile0 << 16);  // tile_dim0
  g1[4] = (int)tile1;                           // tile_dim1 ; tile_dim2=0
  g1[5] = (int)stride0;                         // tensor_dim0_stride[31:0]
  g1[6] = (int)((unsigned long long)stride0 >> 32);  // [47:32] (=0)
  g1[7] = 0;                                    // tensor_dim1_stride unused
  i32x4 z4 = {0, 0, 0, 0};
#if __clang_major__ >= 23
  i32x8 z8 = {0, 0, 0, 0, 0, 0, 0, 0};
  __builtin_amdgcn_tensor_load_to_lds(g0, g1, z4, z4, z8, 0);
#else
  __builtin_amdgcn_tensor_load_to_lds(g0, g1, z4, z4, 0);
#endif
}

// ================= GCN: degree / norm =================
__global__ void deg_init_kernel(float* __restrict__ deg) {
  int n = blockIdx.x * blockDim.x + threadIdx.x;
  if (n < N_NODES) deg[n] = 1.0f;  // self loop
}
__global__ void deg_edge_kernel(const int* __restrict__ dst, float* __restrict__ deg) {
  int e = blockIdx.x * blockDim.x + threadIdx.x;
  if (e < E_EDGES) atomicAdd(&deg[dst[e]], 1.0f);
}
__global__ void dinv_kernel(float* __restrict__ deg) {
  int n = blockIdx.x * blockDim.x + threadIdx.x;
  if (n < N_NODES) deg[n] = rsqrtf(deg[n]);  // deg >= 1 always
}

// ================= GCN feature transform: Y = (relu?)X @ W =================
template <bool RELU>
__global__ void gcn_gemm_kernel(const float* __restrict__ X, const float* __restrict__ W,
                                float* __restrict__ Y, int M, int K) {
  extern __shared__ _Float16 sB[];  // packed B frags: KC x 2 x 32 lanes x 16 halfs
  const int NT = 2;
  int KC = K >> 5;
  int tid = threadIdx.x;
  for (int idx = tid; idx < KC * NT * 32; idx += blockDim.x) {
    int lane = idx & 31, nt = (idx >> 5) % NT, kc = idx / (32 * NT);
    int c = lane & 15, hi = lane >> 4;
    _Float16* dstp = sB + ((size_t)idx << 4);
#pragma unroll
    for (int i = 0; i < 16; ++i) {
      int k = kc * 32 + (i >> 3) * 16 + hi * 8 + (i & 7);  // B[k][n] = W[k][n]
      dstp[i] = (_Float16)W[(size_t)k * GCNH + nt * 16 + c];
    }
  }
  __syncthreads();
  int wave = tid >> 5;
  int row0 = (blockIdx.x * 8 + wave) * 16;
  if (row0 >= M) return;
  int lane = tid & 31;
  v8f acc0 = {}, acc1 = {};
  for (int kc = 0; kc < KC; ++kc) {
    v16h a = load_a_frag<RELU>(X, row0, K, kc * 32);
    v16h b0 = *(const v16h*)(sB + (((size_t)(kc * NT + 0) * 32 + lane) << 4));
    v16h b1 = *(const v16h*)(sB + (((size_t)(kc * NT + 1) * 32 + lane) << 4));
    acc0 = wmma16(a, b0, acc0);
    acc1 = wmma16(a, b1, acc1);
  }
  store_d(Y, row0, GCNH, 0, acc0);
  store_d(Y, row0, GCNH, 16, acc1);
}

// ================= GCN aggregation =================
__global__ void gcn_self_kernel(const float* __restrict__ Hpre, const float* __restrict__ dinv,
                                const float* __restrict__ bias, float* __restrict__ Hagg) {
  int tid = blockIdx.x * blockDim.x + threadIdx.x;
  if (tid >= N_NODES * 8) return;
  int n = tid >> 3, q = (tid & 7) * 4;
  float w = dinv[n] * dinv[n];
  float4 v = *(const float4*)(Hpre + (size_t)n * GCNH + q);
  float4 o;
  o.x = v.x * w + bias[q + 0];
  o.y = v.y * w + bias[q + 1];
  o.z = v.z * w + bias[q + 2];
  o.w = v.w * w + bias[q + 3];
  *(float4*)(Hagg + (size_t)n * GCNH + q) = o;
}

__global__ void gcn_edge_kernel(const float* __restrict__ Hpre, const float* __restrict__ dinv,
                                const int* __restrict__ src, const int* __restrict__ dst,
                                float* __restrict__ Hagg) {
  int tid = blockIdx.x * blockDim.x + threadIdx.x;
  if (tid >= E_EDGES * 8) return;
  int e = tid >> 3, q = (tid & 7) * 4;
  int s = src[e], d = dst[e];
  __builtin_prefetch(Hpre + (size_t)s * GCNH, 0, 0);  // global_prefetch_b8
  float nrm = dinv[s] * dinv[d];
  float4 v = *(const float4*)(Hpre + (size_t)s * GCNH + q);
  float* o = Hagg + (size_t)d * GCNH + q;
  atomicAdd(o + 0, v.x * nrm);
  atomicAdd(o + 1, v.y * nrm);
  atomicAdd(o + 2, v.z * nrm);
  atomicAdd(o + 3, v.w * nrm);
}

// ================= LSTM input-gate precompute: G = X @ Wih^T + (bih+bhh) ====
__global__ void gates_gemm_kernel(const float* __restrict__ X, const float* __restrict__ Wih,
                                  const float* __restrict__ bih, const float* __restrict__ bhh,
                                  float* __restrict__ G, int M, int K) {
  extern __shared__ _Float16 sB[];  // KC x 8 x 32 x 16 halfs
  const int NT = 8;
  int KC = K >> 5;
  int col0 = blockIdx.y * 128;
  int tid = threadIdx.x;
  for (int idx = tid; idx < KC * NT * 32; idx += blockDim.x) {
    int lane = idx & 31, nt = (idx >> 5) % NT, kc = idx / (32 * NT);
    int c = lane & 15, hi = lane >> 4;
    _Float16* dstp = sB + ((size_t)idx << 4);
    int n = col0 + nt * 16 + c;
#pragma unroll
    for (int i = 0; i < 16; ++i) {
      int k = kc * 32 + (i >> 3) * 16 + hi * 8 + (i & 7);  // B[k][n] = Wih[n][k]
      dstp[i] = (_Float16)Wih[(size_t)n * K + k];
    }
  }
  __syncthreads();
  int wave = tid >> 5;
  int row0 = (blockIdx.x * 8 + wave) * 16;
  if (row0 >= M) return;
  int lane = tid & 31, c = lane & 15;
  v8f acc[NT];
#pragma unroll
  for (int nt = 0; nt < NT; ++nt) acc[nt] = (v8f){};
  for (int kc = 0; kc < KC; ++kc) {
    v16h a = load_a_frag<false>(X, row0, K, kc * 32);
#pragma unroll
    for (int nt = 0; nt < NT; ++nt) {
      v16h b = *(const v16h*)(sB + (((size_t)(kc * NT + nt) * 32 + lane) << 4));
      acc[nt] = wmma16(a, b, acc[nt]);
    }
  }
#pragma unroll
  for (int nt = 0; nt < NT; ++nt) {
    int n = col0 + nt * 16 + c;
    float bias = bih[n] + bhh[n];
#pragma unroll
    for (int j = 0; j < 8; ++j) acc[nt][j] += bias;
    store_d(G, row0, 4 * H_, col0 + nt * 16, acc[nt]);
  }
}

// ================= LSTM recurrence (persistent, 1 wave per 16-batch tile) ===
// LDS map (dynamic):  [0,128K)   sW  : Whh packed in B-fragment layout
//                     [128K,132K) sH : 16x128 f16 h tile (next-step A source)
//                     [132K,196K) sG : 2 x (16x512 f32) TDM-filled gate tiles
#define LSTM_SW_BYTES  (4 * 32 * 32 * 16 * 2)          // 131072
#define LSTM_SH_BYTES  (16 * H_ * 2)                   // 4096
#define LSTM_SG_OFF    (LSTM_SW_BYTES + LSTM_SH_BYTES) // 135168
#define LSTM_SG_BYTES  (16 * 512 * 4)                  // 32768 per buffer
#define LSTM_SMEM_TOT  (LSTM_SG_OFF + 2 * LSTM_SG_BYTES)

__global__ void lstm_kernel(const float* __restrict__ G, const float* __restrict__ Whh,
                            float* __restrict__ out, int reverse, int col_off) {
  extern __shared__ char smemraw[];
  _Float16* sW = (_Float16*)smemraw;
  _Float16* sH = (_Float16*)(smemraw + LSTM_SW_BYTES);
  float*    sG = (float*)(smemraw + LSTM_SG_OFF);
  const int KC = 4, NT = 32;
  int lane = threadIdx.x & 31;
  int c = lane & 15, hi = lane >> 4;
  int b0 = blockIdx.x * 16;

  // pack Whh: B[k][n] = Whh[n][k]
  for (int fi = 0; fi < KC * NT; ++fi) {
    int kc = fi / NT, nt = fi % NT;
    _Float16* dstp = sW + (((size_t)fi * 32 + lane) << 4);
#pragma unroll
    for (int i = 0; i < 16; ++i) {
      int k = kc * 32 + (i >> 3) * 16 + hi * 8 + (i & 7);
      dstp[i] = (_Float16)Whh[(size_t)(nt * 16 + c) * H_ + k];
    }
  }
  for (int i = lane; i < 16 * H_; i += 32) sH[i] = (_Float16)0.0f;
  __syncthreads();

  float creg[8][8];
#pragma unroll
  for (int nt = 0; nt < 8; ++nt)
#pragma unroll
    for (int j = 0; j < 8; ++j) creg[nt][j] = 0.0f;

  // prime the TDM pipeline with step 0's gate tile
  {
    int t0 = reverse ? (T_ - 1) : 0;
    tdm_load_gate_tile(LSTM_SG_OFF, G + ((size_t)b0 * T_ + t0) * (4 * H_));
  }

  for (int s = 0; s < T_; ++s) {
    int t = reverse ? (T_ - 1 - s) : s;
    // kick off next step's gate tile into the other buffer
    if (s + 1 < T_) {
      int tn = reverse ? (T_ - 2 - s) : (s + 1);
      tdm_load_gate_tile(LSTM_SG_OFF + ((s + 1) & 1) * LSTM_SG_BYTES,
                         G + ((size_t)b0 * T_ + tn) * (4 * H_));
    }
    // A fragments of h from LDS
    v16h afr[KC];
#pragma unroll
    for (int kc = 0; kc < KC; ++kc) {
      int r = lane & 15;
#pragma unroll
      for (int g = 0; g < 2; ++g) {
        int kb = kc * 32 + g * 16 + hi * 8;
#pragma unroll
        for (int i = 0; i < 8; ++i) afr[kc][g * 8 + i] = sH[r * H_ + kb + i];
      }
    }
    v8f acc[NT];
#pragma unroll
    for (int nt = 0; nt < NT; ++nt) acc[nt] = (v8f){};
#pragma unroll
    for (int kc = 0; kc < KC; ++kc) {
#pragma unroll
      for (int nt = 0; nt < NT; ++nt) {
        v16h b = *(const v16h*)(sW + (((size_t)(kc * NT + nt) * 32 + lane) << 4));
        acc[nt] = wmma16(afr[kc], b, acc[nt]);
      }
    }
    // current tile done (TDM in-order: <=1 outstanding leaves current complete)
    if (s + 1 < T_) __builtin_amdgcn_s_wait_tensorcnt(1);
    else            __builtin_amdgcn_s_wait_tensorcnt(0);

    const float* gt = sG + (s & 1) * (LSTM_SG_BYTES / 4);
    // pointwise LSTM update (gate order i,f,g,o); gates for (b,n) share a lane
#pragma unroll
    for (int nt = 0; nt < 8; ++nt) {
#pragma unroll
      for (int j = 0; j < 8; ++j) {
        int r = j + 8 * hi;               // row within 16-batch tile
        int b = b0 + r;
        int n = nt * 16 + c;
        const float* gr = gt + (size_t)r * (4 * H_);
        float gi = acc[nt][j]      + gr[n];
        float gf = acc[nt + 8][j]  + gr[H_ + n];
        float gg = acc[nt + 16][j] + gr[2 * H_ + n];
        float go = acc[nt + 24][j] + gr[3 * H_ + n];
        float cc = sigmf(gf) * creg[nt][j] + sigmf(gi) * tanh_f(gg);
        creg[nt][j] = cc;
        float hh = sigmf(go) * tanh_f(cc);
        out[((size_t)b * T_ + t) * (2 * H_) + col_off + n] = hh;
        sH[r * H_ + n] = (_Float16)hh;
      }
    }
    __syncthreads();  // fence: sH writes visible before next A-frag rebuild
  }
}

// ================= FC head =================
__global__ void fc_kernel(const float* __restrict__ O, const float* __restrict__ Wfc,
                          const float* __restrict__ bfc, float* __restrict__ Y) {
  int tid = blockIdx.x * blockDim.x + threadIdx.x;
  if (tid >= B_ * NCLS) return;
  int b = tid / NCLS, cl = tid % NCLS;
  const float* v = O + ((size_t)b * T_ + (T_ - 1)) * (2 * H_);
  float s = bfc[cl];
#pragma unroll 4
  for (int k = 0; k < 2 * H_; ++k) s += v[k] * Wfc[(size_t)k * NCLS + cl];
  Y[tid] = s;
}

// ================= orchestration =================
extern "C" void kernel_launch(void* const* d_in, const int* in_sizes, int n_in,
                              void* d_out, int out_size, void* d_ws, size_t ws_size,
                              hipStream_t stream) {
  (void)in_sizes; (void)n_in; (void)out_size; (void)ws_size;
  const float* x     = (const float*)d_in[0];
  const int*   esrc  = (const int*)d_in[1];
  const int*   edst  = (const int*)d_in[2];
  const float* W1 = (const float*)d_in[3];  const float* b1 = (const float*)d_in[4];
  const float* W2 = (const float*)d_in[5];  const float* b2 = (const float*)d_in[6];
  const float* W3 = (const float*)d_in[7];  const float* b3 = (const float*)d_in[8];
  const float* Wih0f = (const float*)d_in[9];  const float* Whh0f = (const float*)d_in[10];
  const float* bih0f = (const float*)d_in[11]; const float* bhh0f = (const float*)d_in[12];
  const float* Wih0b = (const float*)d_in[13]; const float* Whh0b = (const float*)d_in[14];
  const float* bih0b = (const float*)d_in[15]; const float* bhh0b = (const float*)d_in[16];
  const float* Wih1f = (const float*)d_in[17]; const float* Whh1f = (const float*)d_in[18];
  const float* bih1f = (const float*)d_in[19]; const float* bhh1f = (const float*)d_in[20];
  const float* Wih1b = (const float*)d_in[21]; const float* Whh1b = (const float*)d_in[22];
  const float* bih1b = (const float*)d_in[23]; const float* bhh1b = (const float*)d_in[24];
  const float* Wfc = (const float*)d_in[25]; const float* bfc = (const float*)d_in[26];
  float* yout = (float*)d_out;

  // workspace layout (bytes)
  char* ws = (char*)d_ws;
  float* dinv  = (float*)(ws);                                         // N
  float* hA    = (float*)(ws + (size_t)1310720);                       // N*32
  float* hB    = (float*)(ws + (size_t)1310720 + 41943040);            // N*32
  float* gates = (float*)(ws + (size_t)1310720 + 2ull * 41943040);     // B*T*512
  float* out0  = (float*)((char*)gates + (size_t)B_ * T_ * 512 * 4);   // B*T*256
  float* out1  = (float*)((char*)out0  + (size_t)B_ * T_ * 256 * 4);   // B*T*256

  const int M  = N_NODES;       // 327680
  const int MS = B_ * T_;       // 65536

  // ---- GCN normalization ----
  deg_init_kernel<<<(N_NODES + 255) / 256, 256, 0, stream>>>(dinv);
  deg_edge_kernel<<<(E_EDGES + 255) / 256, 256, 0, stream>>>(edst, dinv);
  dinv_kernel<<<(N_NODES + 255) / 256, 256, 0, stream>>>(dinv);

  // ---- GCN layer 1 ----
  gcn_gemm_kernel<false><<<M / 128, 256, 2 * 2 * 32 * 16 * 2, stream>>>(x, W1, hA, M, FIN);
  gcn_self_kernel<<<(N_NODES * 8 + 255) / 256, 256, 0, stream>>>(hA, dinv, b1, hB);
  gcn_edge_kernel<<<(E_EDGES * 8 + 255) / 256, 256, 0, stream>>>(hA, dinv, esrc, edst, hB);
  // ---- GCN layer 2 (ReLU fused into A-load) ----
  gcn_gemm_kernel<true><<<M / 128, 256, 1 * 2 * 32 * 16 * 2, stream>>>(hB, W2, hA, M, GCNH);
  gcn_self_kernel<<<(N_NODES * 8 + 255) / 256, 256, 0, stream>>>(hA, dinv, b2, hB);
  gcn_edge_kernel<<<(E_EDGES * 8 + 255) / 256, 256, 0, stream>>>(hA, dinv, esrc, edst, hB);
  // ---- GCN layer 3 ----
  gcn_gemm_kernel<true><<<M / 128, 256, 1 * 2 * 32 * 16 * 2, stream>>>(hB, W3, hA, M, GCNH);
  gcn_self_kernel<<<(N_NODES * 8 + 255) / 256, 256, 0, stream>>>(hA, dinv, b3, hB);
  gcn_edge_kernel<<<(E_EDGES * 8 + 255) / 256, 256, 0, stream>>>(hA, dinv, esrc, edst, hB);
  // hB reinterpreted as Xseq: (B, T, 160)

  dim3 ggrid(MS / 128, 4);
  const size_t smem160 = (size_t)5 * 8 * 32 * 16 * 2;   // 40 KB
  const size_t smem256 = (size_t)8 * 8 * 32 * 16 * 2;   // 64 KB
  const size_t smemL   = (size_t)LSTM_SMEM_TOT;         // 196 KB

  // ---- BiLSTM layer 0 ----
  gates_gemm_kernel<<<ggrid, 256, smem160, stream>>>(hB, Wih0f, bih0f, bhh0f, gates, MS, LSTM_IN);
  lstm_kernel<<<8, 32, smemL, stream>>>(gates, Whh0f, out0, 0, 0);
  gates_gemm_kernel<<<ggrid, 256, smem160, stream>>>(hB, Wih0b, bih0b, bhh0b, gates, MS, LSTM_IN);
  lstm_kernel<<<8, 32, smemL, stream>>>(gates, Whh0b, out0, 1, H_);

  // ---- BiLSTM layer 1 ----
  gates_gemm_kernel<<<ggrid, 256, smem256, stream>>>(out0, Wih1f, bih1f, bhh1f, gates, MS, 2 * H_);
  lstm_kernel<<<8, 32, smemL, stream>>>(gates, Whh1f, out1, 0, 0);
  gates_gemm_kernel<<<ggrid, 256, smem256, stream>>>(out0, Wih1b, bih1b, bhh1b, gates, MS, 2 * H_);
  lstm_kernel<<<8, 32, smemL, stream>>>(gates, Whh1b, out1, 1, H_);

  // ---- FC head ----
  fc_kernel<<<(B_ * NCLS + 127) / 128, 128, 0, stream>>>(out1, Wfc, bfc, yout);
}
